// Diffusion_Model_Density_73452530696410
// MI455X (gfx1250) — matrix operand order repaired and checked by reading, not compiled, not run on hardware
//
#include <hip/hip_runtime.h>
#include <hip/hip_bf16.h>
#include <math.h>

// ---------------- problem constants (match reference) ----------------
constexpr int NN  = 20000;          // nodes
constexpr int NE  = 320000;         // edges
constexpr int NB  = 8;              // batch
constexpr int NT  = 12;             // T_IN
constexpr int VH  = 32;             // velocity hidden
constexpr int PH  = 64;             // prob hidden
constexpr int ROWS = NE * NB;       // 2,560,000 edge-batch rows
constexpr int NROWS = NN * NB;      // 160,000 node-batch rows
constexpr float LN_EPS = 1e-5f;

typedef __attribute__((ext_vector_type(16))) _Float16 v16h;
typedef __attribute__((ext_vector_type(8)))  float    v8f;

// -------- order-preserving float<->uint for atomicMax on floats --------
__device__ __forceinline__ unsigned fenc(float f) {
    unsigned u = __float_as_uint(f);
    return (u & 0x80000000u) ? ~u : (u | 0x80000000u);
}
__device__ __forceinline__ float fdec(unsigned u) {
    u = (u & 0x80000000u) ? (u ^ 0x80000000u) : ~u;
    return __uint_as_float(u);
}

// ---- DPP-based butterfly add over each 16-lane half of a wave32 ----
// quad_perm(1,0,3,2)=0xB1, quad_perm(2,3,0,1)=0x4E,
// row_half_mirror=0x141, row_mirror=0x140.  All VALU, no LDS, no waits.
template <int CTRL>
__device__ __forceinline__ float dpp_add(float x) {
    int y = __builtin_amdgcn_mov_dpp(__float_as_int(x), CTRL, 0xF, 0xF, true);
    return x + __int_as_float(y);
}
__device__ __forceinline__ float half16_reduce(float x) {
    x = dpp_add<0xB1>(x);    // xor 1 (within quad)
    x = dpp_add<0x4E>(x);    // xor 2 (within quad)
    x = dpp_add<0x141>(x);   // fold 4s -> 8 (row_half_mirror)
    x = dpp_add<0x140>(x);   // fold 8s -> 16 (row_mirror)
    return x;
}

// ============================================================
// K0: zero accumulators + compute attention scalar constants
//   W = sum(attn_w*ln2_g), C = sum(attn_w*ln2_b), wg128
// ============================================================
__global__ void k_init(float* __restrict__ den, float* __restrict__ hbuf,
                       float* __restrict__ pred, unsigned* __restrict__ menc,
                       const float* __restrict__ attn_w,
                       const float* __restrict__ ln2_g,
                       const float* __restrict__ ln2_b,
                       float* __restrict__ consts) {
    int i = blockIdx.x * 256 + threadIdx.x;
    if (i < NROWS) { den[i] = 0.f; hbuf[i] = 0.f; menc[i] = 0u; }
    if (i < NROWS * NT) pred[i] = 0.f;
    if (i == 0) {
        float W = 0.f, C = 0.f;
        for (int k = 0; k < 129; ++k) {
            W += attn_w[k] * ln2_g[k];
            C += attn_w[k] * ln2_b[k];
        }
        consts[0] = W;
        consts[1] = C;
        consts[2] = attn_w[128] * ln2_g[128];
    }
}

// ============================================================
// K1: per (node,b): z = fc_w @ feature; store (s1, s2, pA, pB)
//   s1=sum z, s2=sum z^2, pA=z.wg[0:64], pB=z.wg[64:128]
// ============================================================
__global__ __launch_bounds__(256) void k_nodestats(
        const float* __restrict__ feature, const float* __restrict__ fc_w,
        const float* __restrict__ attn_w, const float* __restrict__ ln2_g,
        float* __restrict__ stats) {
    __shared__ float s_w[PH * NT];
    __shared__ float s_wg[2 * PH];
    for (int i = threadIdx.x; i < PH * NT; i += 256) s_w[i] = fc_w[i];
    for (int i = threadIdx.x; i < 2 * PH; i += 256) s_wg[i] = attn_w[i] * ln2_g[i];
    __syncthreads();

    int idx = blockIdx.x * 256 + threadIdx.x;
    if (idx >= NROWS) return;
    float f[NT];
#pragma unroll
    for (int t = 0; t < NT; ++t) f[t] = feature[idx * NT + t];
    float s1 = 0.f, s2 = 0.f, pA = 0.f, pB = 0.f;
#pragma unroll 4
    for (int h = 0; h < PH; ++h) {
        float z = 0.f;
#pragma unroll
        for (int t = 0; t < NT; ++t) z = fmaf(s_w[h * NT + t], f[t], z);
        s1 += z; s2 = fmaf(z, z, s2);
        pA = fmaf(z, s_wg[h], pA);
        pB = fmaf(z, s_wg[PH + h], pB);
    }
    float4 st; st.x = s1; st.y = s2; st.z = pA; st.w = pB;
    ((float4*)stats)[idx] = st;
}

// ============================================================
// K2: velocity MLPs via WMMA f32_16x16x32_f16.
//   Each wave: 16 rows of [E*B,12] x [12,32] (K zero-padded to 32),
//   hidden 32 = two 16-col N-tiles, two branches => 4 WMMAs.
//   Post: LN(32)+relu+dot+sigmoid per row via DPP half-wave reduce;
//   v = softplus(l3 . [xu,xd]).
// ============================================================
__device__ __forceinline__ void mlp_tail(const v8f& c0, const v8f& c1, int col,
                                         const float* __restrict__ bias,
                                         const float* __restrict__ g,
                                         const float* __restrict__ bb,
                                         const float* __restrict__ w2, float b2,
                                         float* out8) {
#pragma unroll
    for (int r = 0; r < 8; ++r) {
        float x0 = c0[r] + bias[col];
        float x1 = c1[r] + bias[col + 16];
        float s  = half16_reduce(x0 + x1);
        float q  = half16_reduce(x0 * x0 + x1 * x1);
        float mu = s * (1.f / 32.f);
        float rstd = rsqrtf(q * (1.f / 32.f) - mu * mu + LN_EPS);
        float y0 = fmaxf((x0 - mu) * rstd * g[col] + bb[col], 0.f);
        float y1 = fmaxf((x1 - mu) * rstd * g[col + 16] + bb[col + 16], 0.f);
        float d  = half16_reduce(y0 * w2[col] + y1 * w2[col + 16]);
        out8[r] = 1.f / (1.f + expf(-(d + b2)));
    }
}

__global__ __launch_bounds__(256) void k_velocity(
        const float* __restrict__ up, const float* __restrict__ dn,
        const float* __restrict__ l11_w, const float* __restrict__ l11_b,
        const float* __restrict__ ln11_g, const float* __restrict__ ln11_b,
        const float* __restrict__ l12_w, const float* __restrict__ l12_b,
        const float* __restrict__ l21_w, const float* __restrict__ l21_b,
        const float* __restrict__ ln21_g, const float* __restrict__ ln21_b,
        const float* __restrict__ l22_w, const float* __restrict__ l22_b,
        const float* __restrict__ l3_w, const float* __restrict__ l3_b,
        float* __restrict__ v_out) {
    __shared__ float s_w11[VH * NT], s_w21[VH * NT];
    __shared__ float s_b11[VH], s_g11[VH], s_bb11[VH], s_w12[VH];
    __shared__ float s_b21[VH], s_g21[VH], s_bb21[VH], s_w22[VH];
    int tid = threadIdx.x;
    for (int i = tid; i < VH * NT; i += 256) { s_w11[i] = l11_w[i]; s_w21[i] = l21_w[i]; }
    if (tid < VH) {
        s_b11[tid] = l11_b[tid]; s_g11[tid] = ln11_g[tid]; s_bb11[tid] = ln11_b[tid]; s_w12[tid] = l12_w[tid];
        s_b21[tid] = l21_b[tid]; s_g21[tid] = ln21_g[tid]; s_bb21[tid] = ln21_b[tid]; s_w22[tid] = l22_w[tid];
    }
    __syncthreads();

    const int wave = tid >> 5;
    const int lane = tid & 31;
    const int half = lane >> 4;     // 0: K0..7 of A and B; 1: K8..11 of A and B (rest zero pad)
    const int col  = lane & 15;
    const int rowBase = (blockIdx.x * 8 + wave) * 16;   // exact: ROWS % 128 == 0
    const int rLoad = rowBase + col;

    // ---- A fragments (16x32 f16, K>=12 zero-padded); float4 loads (rows are 48B => 16B aligned) ----
    v16h a_u = {}; v16h a_d = {};
    {
        const float4* pu = (const float4*)(up + (size_t)rLoad * NT + half * 8);
        const float4* pd = (const float4*)(dn + (size_t)rLoad * NT + half * 8);
        float4 u0 = pu[0], d0 = pd[0];
        a_u[0] = (_Float16)u0.x; a_u[1] = (_Float16)u0.y; a_u[2] = (_Float16)u0.z; a_u[3] = (_Float16)u0.w;
        a_d[0] = (_Float16)d0.x; a_d[1] = (_Float16)d0.y; a_d[2] = (_Float16)d0.z; a_d[3] = (_Float16)d0.w;
        if (half == 0) {
            float4 u1 = pu[1], d1 = pd[1];
            a_u[4] = (_Float16)u1.x; a_u[5] = (_Float16)u1.y; a_u[6] = (_Float16)u1.z; a_u[7] = (_Float16)u1.w;
            a_d[4] = (_Float16)d1.x; a_d[5] = (_Float16)d1.y; a_d[6] = (_Float16)d1.z; a_d[7] = (_Float16)d1.w;
        }
    }
    // ---- B fragments (32x16 f16): half h elements 0..7 hold K=8h..8h+7 (real K<12), elements 8..15 K=16+8h.. (pad) ----
    v16h bu0 = {}, bu1 = {}, bd0 = {}, bd1 = {};
    {
#pragma unroll
        for (int k = 0; k < 8; ++k) {
            if (half * 8 + k >= NT) continue;
            bu0[k] = (_Float16)s_w11[col * NT + half * 8 + k];
            bu1[k] = (_Float16)s_w11[(col + 16) * NT + half * 8 + k];
            bd0[k] = (_Float16)s_w21[col * NT + half * 8 + k];
            bd1[k] = (_Float16)s_w21[(col + 16) * NT + half * 8 + k];
        }
    }

    v8f z8 = {};
    v8f cu0 = __builtin_amdgcn_wmma_f32_16x16x32_f16(false, a_u, false, bu0, (short)0, z8, false, false);
    v8f cu1 = __builtin_amdgcn_wmma_f32_16x16x32_f16(false, a_u, false, bu1, (short)0, z8, false, false);
    v8f cd0 = __builtin_amdgcn_wmma_f32_16x16x32_f16(false, a_d, false, bd0, (short)0, z8, false, false);
    v8f cd1 = __builtin_amdgcn_wmma_f32_16x16x32_f16(false, a_d, false, bd1, (short)0, z8, false, false);

    float xu[8], xd[8];
    mlp_tail(cu0, cu1, col, s_b11, s_g11, s_bb11, s_w12, l12_b[0], xu);
    mlp_tail(cd0, cd1, col, s_b21, s_g21, s_bb21, s_w22, l22_b[0], xd);

    float w30 = l3_w[0], w31 = l3_w[1], b3 = l3_b[0];
    if (col == 0) {
        float4 o0, o1;
        float* o = &o0.x;
        float vv[8];
#pragma unroll
        for (int r = 0; r < 8; ++r) {
            float t = fmaf(w30, xu[r], fmaf(w31, xd[r], b3));
            vv[r] = log1pf(expf(t));   // softplus
        }
        o0.x = vv[0]; o0.y = vv[1]; o0.z = vv[2]; o0.w = vv[3];
        o1.x = vv[4]; o1.y = vv[5]; o1.z = vv[6]; o1.w = vv[7];
        float4* po = (float4*)(v_out + rowBase + half * 8);   // 32B aligned
        po[0] = o0; po[1] = o1;
        (void)o;
    }
}

// ============================================================
// K3: attention logit via LN decomposition + segment max (by src)
// ============================================================
__global__ __launch_bounds__(256) void k_attn1(
        const int* __restrict__ src, const int* __restrict__ dst,
        const float* __restrict__ v, const float* __restrict__ stats,
        const float* __restrict__ consts,
        float* __restrict__ a_out, unsigned* __restrict__ menc) {
    int r = blockIdx.x * 256 + threadIdx.x;
    if (r >= ROWS) return;
    int e = r >> 3, b = r & 7;
    int s = src[e], d = dst[e];
    float4 ss = ((const float4*)stats)[s * NB + b];
    float4 sd = ((const float4*)stats)[d * NB + b];
    float vv = v[r];
    float W = consts[0], C = consts[1], wg128 = consts[2];
    float s1 = ss.x + sd.x + vv;
    float s2 = ss.y + sd.y + vv * vv;
    float mu = s1 * (1.f / 129.f);
    float var = s2 * (1.f / 129.f) - mu * mu;
    float rstd = rsqrtf(var + LN_EPS);
    float p = ss.z + sd.w + vv * wg128;
    float a = (p - mu * W) * rstd + C;
    a = (a >= 0.f) ? a : 0.01f * a;            // leaky_relu 0.01
    a_out[r] = a;
    atomicMax(&menc[s * NB + b], fenc(a));
}

// ============================================================
// K4: ex = exp(a - max); segment sum of ex (by src). In-place on a_out.
// ============================================================
__global__ __launch_bounds__(256) void k_attn2(
        const int* __restrict__ src, const unsigned* __restrict__ menc,
        float* __restrict__ a_out, float* __restrict__ den) {
    int r = blockIdx.x * 256 + threadIdx.x;
    if (r >= ROWS) return;
    int e = r >> 3, b = r & 7;
    int s = src[e];
    float m = fdec(menc[s * NB + b]);
    float ex = expf(a_out[r] - m);
    a_out[r] = ex;
    atomicAdd(&den[s * NB + b], ex);
}

// ============================================================
// K5: score = ex/den[src]; diffusion message -> h[dst]; weighted
//     feature scatter -> pred[dst] (predicted_outflow).
// ============================================================
__global__ __launch_bounds__(256) void k_message(
        const int* __restrict__ src, const int* __restrict__ dst,
        const float* __restrict__ feature, const float* __restrict__ distance,
        const float* __restrict__ alpha, const float* __restrict__ v,
        const float* __restrict__ ex, const float* __restrict__ den,
        float* __restrict__ hbuf, float* __restrict__ pred) {
    int r = blockIdx.x * 256 + threadIdx.x;
    if (r >= ROWS) return;
    int e = r >> 3, b = r & 7;
    int s = src[e], d = dst[e];
    float score = ex[r] / den[s * NB + b];
    float vv = v[r];
    float Tt = distance[e] / (vv + 1e-5f);
    float al = fminf(fmaxf(alpha[e], 0.001f), 1.0f);
    float F = 1.f / (1.f + al * Tt);
    float Tidx = fminf(fmaxf(rintf(Tt * 0.1f), 0.f), (float)(NT - 1)); // round-half-even
    int n = NT - (int)Tidx;                   // 1..12
    const float* f = feature + (size_t)(s * NB + b) * NT;
    float fl[NT];
#pragma unroll
    for (int t = 0; t < NT; ++t) fl[t] = f[t];
    // sum_t F*(1-F)^(n-1-t)*f_t over t in [0, n-1], ascending powers (no div)
    float g = 1.f - F;
    float acc = 0.f, pw = 1.f;
#pragma unroll
    for (int eo = 0; eo < NT; ++eo) {
        if (eo < n) { acc = fmaf(fl[n - 1 - eo], pw, acc); pw *= g; }
    }
    float msg = score * F * acc;
    atomicAdd(&hbuf[d * NB + b], msg);
    float* pr = pred + (size_t)(d * NB + b) * NT;
#pragma unroll
    for (int t = 0; t < NT; ++t) atomicAdd(&pr[t], fl[t] * score);
}

// ============================================================
// K6: density = pred - feature; bias = relu(d2 . relu(d1 . density)); out = h + bias
// ============================================================
__global__ __launch_bounds__(256) void k_final(
        const float* __restrict__ feature, const float* __restrict__ pred,
        const float* __restrict__ hbuf,
        const float* __restrict__ d1_w, const float* __restrict__ d1_b,
        const float* __restrict__ d2_w, const float* __restrict__ d2_b,
        float* __restrict__ out) {
    __shared__ float s_w1[32 * NT], s_b1[32], s_w2[32];
    for (int i = threadIdx.x; i < 32 * NT; i += 256) s_w1[i] = d1_w[i];
    if (threadIdx.x < 32) { s_b1[threadIdx.x] = d1_b[threadIdx.x]; s_w2[threadIdx.x] = d2_w[threadIdx.x]; }
    __syncthreads();

    int idx = blockIdx.x * 256 + threadIdx.x;
    if (idx >= NROWS) return;
    float dens[NT];
#pragma unroll
    for (int t = 0; t < NT; ++t) dens[t] = pred[idx * NT + t] - feature[idx * NT + t];
    float acc2 = d2_b[0];
#pragma unroll 4
    for (int hh = 0; hh < 32; ++hh) {
        float z = s_b1[hh];
#pragma unroll
        for (int t = 0; t < NT; ++t) z = fmaf(s_w1[hh * NT + t], dens[t], z);
        acc2 = fmaf(fmaxf(z, 0.f), s_w2[hh], acc2);
    }
    out[idx] = hbuf[idx] + fmaxf(acc2, 0.f);
}

// ============================================================
extern "C" void kernel_launch(void* const* d_in, const int* in_sizes, int n_in,
                              void* d_out, int out_size, void* d_ws, size_t ws_size,
                              hipStream_t stream) {
    const float* up       = (const float*)d_in[0];
    const float* dn       = (const float*)d_in[1];
    const float* feature  = (const float*)d_in[2];
    const float* distance = (const float*)d_in[3];
    const int*   src      = (const int*)d_in[4];
    const int*   dst      = (const int*)d_in[5];
    const float* fc_w     = (const float*)d_in[6];
    const float* ln2_g    = (const float*)d_in[7];
    const float* ln2_b    = (const float*)d_in[8];
    const float* attn_w   = (const float*)d_in[9];
    const float* alpha    = (const float*)d_in[10];
    const float* l11_w    = (const float*)d_in[11];
    const float* l11_b    = (const float*)d_in[12];
    const float* ln11_g   = (const float*)d_in[13];
    const float* ln11_b   = (const float*)d_in[14];
    const float* l12_w    = (const float*)d_in[15];
    const float* l12_b    = (const float*)d_in[16];
    const float* l21_w    = (const float*)d_in[17];
    const float* l21_b    = (const float*)d_in[18];
    const float* ln21_g   = (const float*)d_in[19];
    const float* ln21_b   = (const float*)d_in[20];
    const float* l22_w    = (const float*)d_in[21];
    const float* l22_b    = (const float*)d_in[22];
    const float* l3_w     = (const float*)d_in[23];
    const float* l3_b     = (const float*)d_in[24];
    const float* d1_w     = (const float*)d_in[25];
    const float* d1_b     = (const float*)d_in[26];
    const float* d2_w     = (const float*)d_in[27];
    const float* d2_b     = (const float*)d_in[28];
    float* out = (float*)d_out;

    // ---- workspace carve-up (~33 MB) ----
    float* ws      = (float*)d_ws;
    float* consts  = ws;                         // 4
    float* stats   = consts + 4;                 // NROWS*4   (16B aligned)
    float* vbuf    = stats + (size_t)NROWS * 4;  // ROWS
    float* abuf    = vbuf + ROWS;                // ROWS (logits, then ex in-place)
    float* den     = abuf + ROWS;                // NROWS
    float* hbuf    = den + NROWS;                // NROWS
    float* pred    = hbuf + NROWS;               // NROWS*NT
    unsigned* menc = (unsigned*)(pred + (size_t)NROWS * NT); // NROWS

    const int initN = (NROWS * NT + 255) / 256;  // covers all zeroed arrays
    k_init<<<initN, 256, 0, stream>>>(den, hbuf, pred, menc, attn_w, ln2_g, ln2_b, consts);

    k_nodestats<<<(NROWS + 255) / 256, 256, 0, stream>>>(feature, fc_w, attn_w, ln2_g, stats);

    // 8 waves * 16 rows = 128 rows/block; ROWS/128 = 20000 exactly
    k_velocity<<<ROWS / 128, 256, 0, stream>>>(up, dn,
        l11_w, l11_b, ln11_g, ln11_b, l12_w, l12_b,
        l21_w, l21_b, ln21_g, ln21_b, l22_w, l22_b,
        l3_w, l3_b, vbuf);

    const int edgeB = (ROWS + 255) / 256;
    k_attn1<<<edgeB, 256, 0, stream>>>(src, dst, vbuf, stats, consts, abuf, menc);
    k_attn2<<<edgeB, 256, 0, stream>>>(src, menc, abuf, den);
    k_message<<<edgeB, 256, 0, stream>>>(src, dst, feature, distance, alpha, vbuf,
                                         abuf, den, hbuf, pred);
    k_final<<<(NROWS + 255) / 256, 256, 0, stream>>>(feature, pred, hbuf,
                                                     d1_w, d1_b, d2_w, d2_b, out);
}
